// CrfRnnLayer_45715631898812
// MI455X (gfx1250) — compile-verified, matrix operand order checked by reading, not run on hardware
//
#include <hip/hip_runtime.h>
#include <hip/hip_bf16.h>

// CRF-RNN forward for MI455X (gfx1250, wave32, WMMA).
// - Gaussian kernel entries generated on the fly as 2^(hi+hj+gi.gj) where
//   g = feature * sqrt(log2 e), h = -0.5*|g|^2 (precomputed): FMA chain +
//   v_exp_f32 directly (no scaling mul), feeding v_wmma_f32_16x16x32_f16.
// - Feature tiles double-buffered in LDS via GLOBAL_LOAD_ASYNC_TO_LDS
//   (ASYNCcnt), prefetching tile k+1 while computing tile k.

typedef _Float16 v16h __attribute__((ext_vector_type(16)));
typedef _Float16 v8h  __attribute__((ext_vector_type(8)));
typedef float    v8f  __attribute__((ext_vector_type(8)));
typedef int      v2i  __attribute__((vector_size(2 * sizeof(int))));

#define HH 96
#define WW 96
#define CC 21
#define CP 32
#define NN 9216
#define BB 2

#define INV_TG (1.0f / 3.0f)    // 1/theta_gamma (spatial kernel)
#define INV_TA (1.0f / 160.0f)  // 1/theta_alpha (bilateral position)
#define INV_TB (1.0f / 3.0f)    // 1/theta_beta  (bilateral rgb)
#define SQL    1.2011224087864498f  // sqrt(log2(e))

#if __has_builtin(__builtin_amdgcn_global_load_async_to_lds_b32) && \
    __has_builtin(__builtin_amdgcn_global_load_async_to_lds_b64)
#define HAVE_ASYNC_LDS 1
#else
#define HAVE_ASYNC_LDS 0
#endif

#define AS_GLB __attribute__((address_space(1)))
#define AS_LDS __attribute__((address_space(3)))

// ---------------------------------------------------------------- prep
// fs4[n] = {gy, gx, h, 0}         (spatial, g = pos/tg * SQL, h = -0.5|g|^2)
// fb8[b][n] = {g0..g4, h, 0, 0}   (bilateral, g = {pos/ta, rgb/tb} * SQL)
__global__ __launch_bounds__(256) void prep_kernel(const float* __restrict__ rgb,
                                                   float* __restrict__ fs4,
                                                   float* __restrict__ fb8) {
    int n = blockIdx.x * 256 + threadIdx.x;
    if (n >= NN) return;
    float y = (float)(n / WW);
    float x = (float)(n % WW);
    float gy = y * (INV_TG * SQL), gx = x * (INV_TG * SQL);
    fs4[4 * n + 0] = gy;
    fs4[4 * n + 1] = gx;
    fs4[4 * n + 2] = -0.5f * (gy * gy + gx * gx);
    fs4[4 * n + 3] = 0.0f;
#pragma unroll
    for (int b = 0; b < BB; ++b) {
        const float* rp = rgb + ((size_t)b * NN + n) * 3;
        float* fp = fb8 + ((size_t)b * NN + n) * 8;
        float g0 = y * (INV_TA * SQL);
        float g1 = x * (INV_TA * SQL);
        float g2 = rp[0] * (INV_TB * SQL);
        float g3 = rp[1] * (INV_TB * SQL);
        float g4 = rp[2] * (INV_TB * SQL);
        fp[0] = g0; fp[1] = g1; fp[2] = g2; fp[3] = g3; fp[4] = g4;
        fp[5] = -0.5f * (g0 * g0 + g1 * g1 + g2 * g2 + g3 * g3 + g4 * g4);
        fp[6] = 0.0f; fp[7] = 0.0f;
    }
}

// ---------------------------------------------------------------- norms
// rs[j] = 1 / sum_i Ks[i,j] ; rb[b][j] = 1 / sum_i Kb[i,j]
__global__ __launch_bounds__(256) void norms_kernel(const float* __restrict__ fs4,
                                                    const float* __restrict__ fb8,
                                                    float* __restrict__ rs,
                                                    float* __restrict__ rb) {
    __shared__ float lfs[64 * 4];        // 64 i-rows spatial {gy,gx,h,0}
    __shared__ float lfb[2 * 64 * 8];    // 64 i-rows x 2 batches bilateral
    const int t = threadIdx.x;
    const int j = blockIdx.x * 256 + t;

    const float4 sj  = *(const float4*)(fs4 + 4 * j);
    const float4 b04 = *(const float4*)(fb8 + (size_t)j * 8);
    const float2 b02 = *(const float2*)(fb8 + (size_t)j * 8 + 4);
    const float4 b14 = *(const float4*)(fb8 + ((size_t)NN + j) * 8);
    const float2 b12 = *(const float2*)(fb8 + ((size_t)NN + j) * 8 + 4);

    float ss = 0.0f, s0 = 0.0f, s1 = 0.0f;
    for (int i0 = 0; i0 < NN; i0 += 64) {
        __syncthreads();
        lfs[t]       = fs4[i0 * 4 + t];                       // 256 floats
        lfb[t]       = fb8[(size_t)i0 * 8 + t];               // batch 0: 512
        lfb[t + 256] = fb8[(size_t)i0 * 8 + 256 + t];
        lfb[t + 512] = fb8[((size_t)NN + i0) * 8 + t];        // batch 1: 512
        lfb[t + 768] = fb8[((size_t)NN + i0) * 8 + 256 + t];
        __syncthreads();
#pragma unroll 4
        for (int ii = 0; ii < 64; ++ii) {
            const float4 si = *(const float4*)&lfs[ii * 4];
            float ts = fmaf(si.x, sj.x, si.z + sj.z);
            ts = fmaf(si.y, sj.y, ts);
            ss += __builtin_amdgcn_exp2f(ts);
            const float4 a4 = *(const float4*)&lfb[ii * 8];
            const float2 a2 = *(const float2*)&lfb[ii * 8 + 4];
            float t0 = fmaf(a4.x, b04.x, a2.y + b02.y);
            t0 = fmaf(a4.y, b04.y, t0);
            t0 = fmaf(a4.z, b04.z, t0);
            t0 = fmaf(a4.w, b04.w, t0);
            t0 = fmaf(a2.x, b02.x, t0);
            s0 += __builtin_amdgcn_exp2f(t0);
            const float4 c4 = *(const float4*)&lfb[512 + ii * 8];
            const float2 c2 = *(const float2*)&lfb[512 + ii * 8 + 4];
            float t1 = fmaf(c4.x, b14.x, c2.y + b12.y);
            t1 = fmaf(c4.y, b14.y, t1);
            t1 = fmaf(c4.z, b14.z, t1);
            t1 = fmaf(c4.w, b14.w, t1);
            t1 = fmaf(c2.x, b12.x, t1);
            s1 += __builtin_amdgcn_exp2f(t1);
        }
    }
    rs[j] = 1.0f / ss;
    rb[j] = 1.0f / s0;
    rb[NN + j] = 1.0f / s1;
}

// ---------------------------------------------------------------- softmax init
__global__ __launch_bounds__(256) void softmax_init_kernel(const float* __restrict__ u,
                                                           _Float16* __restrict__ smh) {
    const int b = blockIdx.y;
    const int n = blockIdx.x * 256 + threadIdx.x;
    const float* up = u + ((size_t)b * NN + n) * CC;
    float q[CC];
    float m = -1e30f;
#pragma unroll
    for (int c = 0; c < CC; ++c) { q[c] = up[c]; m = fmaxf(m, q[c]); }
    float s = 0.0f;
#pragma unroll
    for (int c = 0; c < CC; ++c) { q[c] = __expf(q[c] - m); s += q[c]; }
    float inv = 1.0f / s;
#pragma unroll
    for (int c = 0; c < CP; ++c)
        smh[((size_t)b * CP + c) * NN + n] = (c < CC) ? (_Float16)(q[c] * inv) : (_Float16)0.0f;
}

// ---------------------------------------------------------------- fused filter
// sp[b][c][j] = (sum_i sm[c,i]*Ks[i,j]) / sn[j] ; bl likewise with Kb.
__global__ __launch_bounds__(128) void filter_kernel(const float* __restrict__ fs4,
                                                     const float* __restrict__ fb8,
                                                     const float* __restrict__ rs,
                                                     const float* __restrict__ rb,
                                                     const _Float16* __restrict__ smh,
                                                     float* __restrict__ sp,
                                                     float* __restrict__ bl) {
    const int b    = blockIdx.y;
    const int t    = threadIdx.x;
    const int lane = t & 31;
    const int wave = t >> 5;
    const int col  = lane & 15;
    const int j    = blockIdx.x * 64 + wave * 16 + col;   // this lane's output column
    const int koff = (lane < 16) ? 0 : 8;                 // K sub-block per ISA layout

    // column-j features
    const float4 sj  = *(const float4*)(fs4 + 4 * j);
    const float4 bj4 = *(const float4*)(fb8 + ((size_t)b * NN + j) * 8);
    const float2 bj2 = *(const float2*)(fb8 + ((size_t)b * NN + j) * 8 + 4);

    __shared__ float lfs[2][32 * 4];   // double-buffered i-tile, spatial
    __shared__ float lfb[2][32 * 8];   // double-buffered i-tile, bilateral

    v8f acc_s0 = {0.f,0.f,0.f,0.f,0.f,0.f,0.f,0.f};
    v8f acc_s1 = acc_s0, acc_b0 = acc_s0, acc_b1 = acc_s0;

    const _Float16* smb = smh + (size_t)b * CP * NN;
    const float* gfb = fb8 + (size_t)b * NN * 8;
    const int KSTEPS = NN / 32;   // 288

#if HAVE_ASYNC_LDS
#define STAGE_ASYNC(buf, i0)                                                         \
    do {                                                                             \
        __builtin_amdgcn_global_load_async_to_lds_b32(                               \
            (AS_GLB int*)(fs4 + (i0) * 4 + t),                                       \
            (AS_LDS int*)&lfs[buf][t], 0, 0);                                        \
        __builtin_amdgcn_global_load_async_to_lds_b64(                               \
            (AS_GLB v2i*)(gfb + (size_t)(i0) * 8 + 2 * t),                           \
            (AS_LDS v2i*)&lfb[buf][2 * t], 0, 0);                                    \
    } while (0)

    STAGE_ASYNC(0, 0);
#endif

    for (int k = 0; k < KSTEPS; ++k) {
        const int cur = k & 1;
        const int i0  = k * 32;
#if HAVE_ASYNC_LDS
        if (k + 1 < KSTEPS) {
            STAGE_ASYNC(cur ^ 1, (k + 1) * 32);
            asm volatile("s_wait_asynccnt 0x2" ::: "memory");  // tile k resident (in-order)
        } else {
            asm volatile("s_wait_asynccnt 0x0" ::: "memory");
        }
        __syncthreads();
#else
        __syncthreads();
        if (t < 128) lfs[cur][t] = fs4[i0 * 4 + t];
        {
            const float* src = gfb + (size_t)i0 * 8;
            lfb[cur][t] = src[t];
            lfb[cur][t + 128] = src[t + 128];
        }
        __syncthreads();
#endif
        const float* Ls = &lfs[cur][0];
        const float* Lb = &lfb[cur][0];

        // ---- generate B fragments (kernel tiles): FMA chain + v_exp_f32 -> f16
        v16h bsf, bbf;
#pragma unroll
        for (int e = 0; e < 16; ++e) {
            const int ii = koff + ((e < 8) ? e : (8 + e)); // K index in [0,32)
            const float4 si = *(const float4*)&Ls[ii * 4];
            float ts = fmaf(si.x, sj.x, si.z + sj.z);
            ts = fmaf(si.y, sj.y, ts);
            bsf[e] = (_Float16)__builtin_amdgcn_exp2f(ts);
            const float4 bi4 = *(const float4*)&Lb[ii * 8];
            const float2 bi2 = *(const float2*)&Lb[ii * 8 + 4];
            float tb = fmaf(bi4.x, bj4.x, bi2.y + bj2.y);
            tb = fmaf(bi4.y, bj4.y, tb);
            tb = fmaf(bi4.z, bj4.z, tb);
            tb = fmaf(bi4.w, bj4.w, tb);
            tb = fmaf(bi2.x, bj2.x, tb);
            bbf[e] = (_Float16)__builtin_amdgcn_exp2f(tb);
        }

        // ---- A fragments: sm rows (M-tile 0: c=col, M-tile 1: c=16+col)
        const _Float16* a0p = smb + (size_t)col * NN + i0 + koff;
        const _Float16* a1p = smb + ((size_t)(16 + col)) * NN + i0 + koff;
        v8h a0lo = *(const v8h*)a0p, a0hi = *(const v8h*)(a0p + 16);
        v8h a1lo = *(const v8h*)a1p, a1hi = *(const v8h*)(a1p + 16);
        v16h a0, a1;
#pragma unroll
        for (int e = 0; e < 8; ++e) {
            a0[e] = a0lo[e]; a0[8 + e] = a0hi[e];
            a1[e] = a1lo[e]; a1[8 + e] = a1hi[e];
        }

        acc_s0 = __builtin_amdgcn_wmma_f32_16x16x32_f16(false, a0, false, bsf, (short)0, acc_s0, false, false);
        acc_s1 = __builtin_amdgcn_wmma_f32_16x16x32_f16(false, a1, false, bsf, (short)0, acc_s1, false, false);
        acc_b0 = __builtin_amdgcn_wmma_f32_16x16x32_f16(false, a0, false, bbf, (short)0, acc_b0, false, false);
        acc_b1 = __builtin_amdgcn_wmma_f32_16x16x32_f16(false, a1, false, bbf, (short)0, acc_b1, false, false);

        __syncthreads();   // protect LDS buffer reuse (next-next tile's async writes)
    }

    // ---- epilogue: multiply by reciprocal norms, store rows < C
    const float rsj = rs[j];
    const float rbj = rb[(size_t)b * NN + j];
    const int rbase = (lane < 16) ? 0 : 8;
#pragma unroll
    for (int r = 0; r < 8; ++r) {
        int row0 = r + rbase;          // M-tile 0
        int row1 = 16 + r + rbase;     // M-tile 1
        if (row0 < CC) {
            sp[((size_t)b * CP + row0) * NN + j] = acc_s0[r] * rsj;
            bl[((size_t)b * CP + row0) * NN + j] = acc_b0[r] * rbj;
        }
        if (row1 < CC) {
            sp[((size_t)b * CP + row1) * NN + j] = acc_s1[r] * rsj;
            bl[((size_t)b * CP + row1) * NN + j] = acc_b1[r] * rbj;
        }
    }
}

// ---------------------------------------------------------------- update
// msg = Ws@sp + Wb@bl ; q = u - M@msg ; write q to d_out [b][n][c];
// sm_h = softmax_c(q) for the next iteration.
__global__ __launch_bounds__(256) void update_kernel(const float* __restrict__ u,
                                                     const float* __restrict__ Ws,
                                                     const float* __restrict__ Wb,
                                                     const float* __restrict__ Mm,
                                                     const float* __restrict__ sp,
                                                     const float* __restrict__ bl,
                                                     float* __restrict__ qout,
                                                     _Float16* __restrict__ smh) {
    __shared__ float sWs[CC * CC], sWb[CC * CC], sM[CC * CC];
    for (int idx = threadIdx.x; idx < CC * CC; idx += 256) {
        sWs[idx] = Ws[idx]; sWb[idx] = Wb[idx]; sM[idx] = Mm[idx];
    }
    __syncthreads();

    const int b = blockIdx.y;
    const int n = blockIdx.x * 256 + threadIdx.x;

    float spv[CC], blv[CC];
#pragma unroll
    for (int c = 0; c < CC; ++c) {
        spv[c] = sp[((size_t)b * CP + c) * NN + n];
        blv[c] = bl[((size_t)b * CP + c) * NN + n];
    }
    float msg[CC];
#pragma unroll
    for (int c = 0; c < CC; ++c) {
        float s = 0.0f;
#pragma unroll
        for (int k = 0; k < CC; ++k)
            s += sWs[c * CC + k] * spv[k] + sWb[c * CC + k] * blv[k];
        msg[c] = s;
    }
    const float* up = u + ((size_t)b * NN + n) * CC;
    float* qp = qout + ((size_t)b * NN + n) * CC;
    float q[CC];
    float m = -1e30f;
#pragma unroll
    for (int c = 0; c < CC; ++c) {
        float s = 0.0f;
#pragma unroll
        for (int k = 0; k < CC; ++k) s += sM[c * CC + k] * msg[k];
        q[c] = up[c] - s;
        qp[c] = q[c];
        m = fmaxf(m, q[c]);
    }
    float s = 0.0f;
#pragma unroll
    for (int c = 0; c < CC; ++c) { q[c] = __expf(q[c] - m); s += q[c]; }
    float inv = 1.0f / s;
#pragma unroll
    for (int c = 0; c < CP; ++c)
        smh[((size_t)b * CP + c) * NN + n] = (c < CC) ? (_Float16)(q[c] * inv) : (_Float16)0.0f;
}

// ---------------------------------------------------------------- launch
extern "C" void kernel_launch(void* const* d_in, const int* in_sizes, int n_in,
                              void* d_out, int out_size, void* d_ws, size_t ws_size,
                              hipStream_t stream) {
    const float* unary = (const float*)d_in[0];   // [B,H,W,C] == [b][n][c]
    const float* rgb   = (const float*)d_in[1];   // [B,H,W,3]
    const float* Ws    = (const float*)d_in[2];   // [C,C]
    const float* Wb    = (const float*)d_in[3];   // [C,C]
    const float* Mm    = (const float*)d_in[4];   // [C,C]
    float* out = (float*)d_out;

    // workspace carve-up (all 256B aligned by construction)
    char* w = (char*)d_ws;
    float*    fs4 = (float*)w;            w += (size_t)NN * 4 * 4;        // 144 KB
    float*    fb8 = (float*)w;            w += (size_t)BB * NN * 8 * 4;   // 576 KB
    float*    rs  = (float*)w;            w += (size_t)NN * 4;            //  36 KB
    float*    rb  = (float*)w;            w += (size_t)BB * NN * 4;       //  72 KB
    _Float16* smh = (_Float16*)w;         w += (size_t)BB * CP * NN * 2;  // 1.2 MB
    float*    sp  = (float*)w;            w += (size_t)BB * CP * NN * 4;  // 2.4 MB
    float*    bl  = (float*)w;            w += (size_t)BB * CP * NN * 4;  // 2.4 MB
    (void)ws_size; (void)in_sizes; (void)n_in; (void)out_size;

    const int NB = NN / 256;  // 36

    prep_kernel<<<NB, 256, 0, stream>>>(rgb, fs4, fb8);
    norms_kernel<<<NB, 256, 0, stream>>>(fs4, fb8, rs, rb);
    softmax_init_kernel<<<dim3(NB, BB), 256, 0, stream>>>(unary, smh);

    for (int it = 0; it < 5; ++it) {
        filter_kernel<<<dim3(NN / 64, BB), 128, 0, stream>>>(fs4, fb8, rs, rb, smh, sp, bl);
        update_kernel<<<dim3(NB, BB), 256, 0, stream>>>(unary, Ws, Wb, Mm, sp, bl, out, smh);
    }
}
